// TtLlamaAttention_optimized_77464030151089
// MI455X (gfx1250) — compile-verified, matrix operand order checked by reading, not run on hardware
//
#include <hip/hip_runtime.h>
#include <hip/hip_bf16.h>
#include <stdint.h>

#define N_HEADS  64
#define N_KV     8
#define HD       128
#define HIDDEN   8192
#define SEQ      2048
#define CACHE_L  2176
#define QKV_N    10240
#define SM_SCALE 0.08838834764831845f   // 1/sqrt(128)

typedef __attribute__((ext_vector_type(16))) __bf16 v16bf;
typedef __attribute__((ext_vector_type(8)))  float  v8f;

__device__ __forceinline__ unsigned short f2bf(float f) {
  unsigned int u = __float_as_uint(f);
  return (unsigned short)((u + 0x7FFFu + ((u >> 16) & 1u)) >> 16);  // RNE
}

__device__ __forceinline__ v16bf ldfrag(const unsigned short* p) {
  return *(const v16bf*)p;  // 32B contiguous -> 2x b128
}

__device__ __forceinline__ v8f wmma_bf16(v16bf a, v16bf b, v8f c) {
  return __builtin_amdgcn_wmma_f32_16x16x32_bf16(
      false, a, false, b, (short)0, c, false, false);
}

// async copy of one 16B chunk, global -> LDS (ASYNCcnt-tracked)
__device__ __forceinline__ void async_cp16(uint32_t lds_byte, const void* gptr) {
  uint64_t ga = (uint64_t)(uintptr_t)gptr;
  asm volatile("global_load_async_to_lds_b128 %0, %1, off"
               :: "v"(lds_byte), "v"(ga) : "memory");
}

// ---------------------------------------------------------------------------
// elementwise f32 -> bf16
// ---------------------------------------------------------------------------
__global__ void cvt_bf16(const float* __restrict__ src,
                         unsigned short* __restrict__ dst, int n) {
  int i = blockIdx.x * blockDim.x + threadIdx.x;
  if (i < n) dst[i] = f2bf(src[i]);
}

// ---------------------------------------------------------------------------
// transpose-convert: W[K][N] f32 -> Wt[N][K] bf16 (32x32 tiles through LDS)
// ---------------------------------------------------------------------------
__global__ __launch_bounds__(256)
void cvt_transpose_bf16(const float* __restrict__ W,
                        unsigned short* __restrict__ Wt, int K, int N) {
  __shared__ unsigned short tile[32][33];
  const int kb = blockIdx.y * 32, nb = blockIdx.x * 32;
  const int tx = threadIdx.x & 31, ty = threadIdx.x >> 5;  // 32 x 8
#pragma unroll
  for (int r = 0; r < 4; ++r) {
    int k = kb + ty + r * 8;
    tile[ty + r * 8][tx] = f2bf(W[(size_t)k * N + nb + tx]);
  }
  __syncthreads();
#pragma unroll
  for (int r = 0; r < 4; ++r) {
    int n = nb + ty + r * 8;
    Wt[(size_t)n * K + kb + tx] = tile[tx][ty + r * 8];
  }
}

// ---------------------------------------------------------------------------
// C[M][N] f32 = A[M][K] bf16 * Bt[N][K] bf16   (both K-major, fragment-ready)
// BM=BN=128, BK=32; 8 waves; tiles staged via async global->LDS, double-buffered
// ---------------------------------------------------------------------------
__global__ __launch_bounds__(256)
void gemm_bf16_async(const unsigned short* __restrict__ A,
                     const unsigned short* __restrict__ Bt,
                     float* __restrict__ C, int M, int N, int K) {
  __shared__ __align__(32) unsigned short As[2][128 * 32];
  __shared__ __align__(32) unsigned short Bs[2][128 * 32];
  const int tid   = threadIdx.x;
  const int lane  = tid & 31;
  const int wave  = tid >> 5;
  const int wm    = wave & 3;
  const int wn    = wave >> 2;
  const int m0    = blockIdx.y * 128;
  const int n0    = blockIdx.x * 128;
  const int row0  = wm * 32;
  const int col0  = wn * 64;
  const int l16   = lane & 15;
  const int lhalf = lane >> 4;

  v8f acc[2][4] = {};

  // each thread copies 2 A-chunks + 2 B-chunks of 16B per tile (512 chunks/tile)
  auto issue_tile = [&](int buf, int k0) {
    uint32_t lA = (uint32_t)(uintptr_t)(&As[buf][0]);
    uint32_t lB = (uint32_t)(uintptr_t)(&Bs[buf][0]);
#pragma unroll
    for (int i = 0; i < 2; ++i) {
      int c = tid + i * 256;
      int row = c >> 2, part = c & 3;
      async_cp16(lA + row * 64 + part * 16,
                 (const char*)(A + (size_t)(m0 + row) * K + k0) + part * 16);
      async_cp16(lB + row * 64 + part * 16,
                 (const char*)(Bt + (size_t)(n0 + row) * K + k0) + part * 16);
    }
  };

  const int T = K >> 5;
  issue_tile(0, 0);
  for (int kt = 0; kt < T; ++kt) {
    const int cur = kt & 1;
    if (kt + 1 < T) {
      issue_tile(1 - cur, (kt + 1) * 32);
      // leave only the 4 just-issued (next tile) in flight -> current tile done
      asm volatile("s_wait_asynccnt 0x4" ::: "memory");
    } else {
      asm volatile("s_wait_asynccnt 0x0" ::: "memory");
    }
    __syncthreads();

    v16bf af[2], bfr[4];
#pragma unroll
    for (int r = 0; r < 2; ++r)
      af[r] = ldfrag(&As[cur][(row0 + 16 * r + l16) * 32 + lhalf * 16]);
#pragma unroll
    for (int c = 0; c < 4; ++c)
      bfr[c] = ldfrag(&Bs[cur][(col0 + 16 * c + l16) * 32 + lhalf * 16]);
#pragma unroll
    for (int r = 0; r < 2; ++r)
#pragma unroll
      for (int c = 0; c < 4; ++c)
        acc[r][c] = wmma_bf16(af[r], bfr[c], acc[r][c]);
    __syncthreads();  // all reads of buf[cur] done before it is refilled
  }

#pragma unroll
  for (int r = 0; r < 2; ++r)
#pragma unroll
    for (int c = 0; c < 4; ++c) {
      int col = n0 + col0 + 16 * c + l16;
#pragma unroll
      for (int i = 0; i < 8; ++i) {
        int row = m0 + row0 + 16 * r + 8 * lhalf + i;
        C[(size_t)row * N + col] = acc[r][c][i];
      }
    }
}

// ---------------------------------------------------------------------------
// seed bf16 caches from f32 inputs: K keeps [kvh][pos][d]; V transposed to
// [kvh][d][pos] so P*V B-fragments are contiguous global loads.
// ---------------------------------------------------------------------------
__global__ void cvt_cache(const float* __restrict__ ck, const float* __restrict__ cv,
                          unsigned short* __restrict__ kc,
                          unsigned short* __restrict__ vct, int n) {
  int i = blockIdx.x * blockDim.x + threadIdx.x;
  if (i < n) {
    kc[i] = f2bf(ck[i]);
    int kvh = i / (CACHE_L * HD);
    int rem = i - kvh * (CACHE_L * HD);
    int pos = rem >> 7;          // /HD
    int d   = rem & (HD - 1);
    vct[((size_t)kvh * HD + d) * CACHE_L + pos] = f2bf(cv[i]);
  }
}

// ---------------------------------------------------------------------------
// RoPE + scatter: q -> bf16 [h][s][128]; k -> [kvh][pos][128]; v -> [kvh][d][pos]
// blockIdx.x = head 0..79, blockIdx.y = s, threadIdx.x = d pair 0..63
// ---------------------------------------------------------------------------
__global__ void rope_scatter(const float* __restrict__ qkv,
                             unsigned short* __restrict__ qbf,
                             unsigned short* __restrict__ kc,
                             unsigned short* __restrict__ vct,
                             const int* __restrict__ spp) {
  const int h = blockIdx.x;
  const int s = blockIdx.y;
  const int d = threadIdx.x;
  const int pos = *spp + s;
  float inv_freq = __powf(10000.0f, -((float)(2 * d)) / 128.0f);
  float ang = (float)pos * inv_freq;
  float cs = __cosf(ang), sn = __sinf(ang);
  const float* row = qkv + (size_t)s * QKV_N;
  if (h < 64) {
    const float* p = row + h * HD;
    float x0 = p[d], x1 = p[d + 64];
    unsigned short* q = qbf + ((size_t)h * SEQ + s) * HD;
    q[d]      = f2bf(x0 * cs - x1 * sn);
    q[d + 64] = f2bf(x1 * cs + x0 * sn);
  } else if (h < 72) {
    int kh = h - 64;
    const float* p = row + HIDDEN + kh * HD;
    float x0 = p[d], x1 = p[d + 64];
    unsigned short* kd = kc + ((size_t)kh * CACHE_L + pos) * HD;
    kd[d]      = f2bf(x0 * cs - x1 * sn);
    kd[d + 64] = f2bf(x1 * cs + x0 * sn);
  } else {
    int vh = h - 72;
    const float* p = row + HIDDEN + N_KV * HD + vh * HD;
    vct[((size_t)vh * HD + d)      * CACHE_L + pos] = f2bf(p[d]);
    vct[((size_t)vh * HD + d + 64) * CACHE_L + pos] = f2bf(p[d + 64]);
  }
}

// ---------------------------------------------------------------------------
// Flash attention, fully wave-independent (no block barriers): each wave owns
// 16 q rows; K and V fragments are direct contiguous global loads; P goes
// through a tiny per-wave LDS transpose. Output written as bf16 [s][8192].
// ---------------------------------------------------------------------------
__global__ __launch_bounds__(128)
void flash_attn(const unsigned short* __restrict__ qbf,
                const unsigned short* __restrict__ kc,
                const unsigned short* __restrict__ vct,
                unsigned short* __restrict__ attnb,
                const int* __restrict__ spp) {
  __shared__ __align__(32) unsigned short Ps[4 * 16 * 32];  // per-wave region
  const int tid   = threadIdx.x;
  const int lane  = tid & 31;
  const int wave  = tid >> 5;
  const int l16   = lane & 15;
  const int lhalf = lane >> 4;
  const int h     = blockIdx.y;
  const int kvh   = h >> 3;
  const int qrow0 = blockIdx.x * 64 + wave * 16;
  const int sp0   = *spp;
  const int ktile = (sp0 + qrow0 + 16 + 31) >> 5;   // per-wave causal bound

  v16bf qf[4];
#pragma unroll
  for (int ch = 0; ch < 4; ++ch)
    qf[ch] = ldfrag(qbf + ((size_t)h * SEQ + qrow0 + l16) * HD + ch * 32 + lhalf * 16);

  v8f o[8] = {};
  float mi[8], li[8];
#pragma unroll
  for (int i = 0; i < 8; ++i) { mi[i] = -3.0e38f; li[i] = 0.f; }

  unsigned short* pw = Ps + wave * (16 * 32);
  const unsigned short* kbase = kc + (size_t)kvh * CACHE_L * HD;
  const unsigned short* vbase = vct + (size_t)kvh * HD * CACHE_L;

  for (int kt = 0; kt < ktile; ++kt) {
    const int k0 = kt * 32;

    // ---- S = Q K^T : two 16x16 tiles, contraction over 4x32 d-chunks
    v8f sacc[2] = {};
#pragma unroll
    for (int t = 0; t < 2; ++t)
#pragma unroll
      for (int ch = 0; ch < 4; ++ch) {
        v16bf kf = ldfrag(kbase + (size_t)(k0 + 16 * t + l16) * HD + ch * 32 + lhalf * 16);
        sacc[t] = wmma_bf16(qf[ch], kf, sacc[t]);
      }

    // ---- mask + online softmax (row spans 16 lanes)
    float p[2][8];
#pragma unroll
    for (int i = 0; i < 8; ++i) {
      int qpos = sp0 + qrow0 + 8 * lhalf + i;
      float s0 = sacc[0][i] * SM_SCALE;
      float s1 = sacc[1][i] * SM_SCALE;
      if (k0 + l16 > qpos)      s0 = -3.0e38f;
      if (k0 + 16 + l16 > qpos) s1 = -3.0e38f;
      float rm = fmaxf(s0, s1);
#pragma unroll
      for (int msk = 1; msk < 16; msk <<= 1)
        rm = fmaxf(rm, __shfl_xor(rm, msk, 32));
      float mnew  = fmaxf(mi[i], rm);
      float alpha = __expf(mi[i] - mnew);
      float p0 = __expf(s0 - mnew);
      float p1 = __expf(s1 - mnew);
      float rs = p0 + p1;
#pragma unroll
      for (int msk = 1; msk < 16; msk <<= 1)
        rs += __shfl_xor(rs, msk, 32);
      li[i] = li[i] * alpha + rs;
      mi[i] = mnew;
#pragma unroll
      for (int c = 0; c < 8; ++c) o[c][i] *= alpha;
      p[0][i] = p0; p[1][i] = p1;
    }

    // ---- per-wave P transpose through LDS, then O += P V
#pragma unroll
    for (int t = 0; t < 2; ++t)
#pragma unroll
      for (int i = 0; i < 8; ++i)
        pw[(8 * lhalf + i) * 32 + 16 * t + l16] = f2bf(p[t][i]);
    asm volatile("s_wait_dscnt 0" ::: "memory");  // same-wave DS RAW
    v16bf pa = ldfrag(pw + l16 * 32 + lhalf * 16);
#pragma unroll
    for (int c = 0; c < 8; ++c) {
      // V^T fragment: 16 consecutive keys of column d -> contiguous 32B
      v16bf vf = ldfrag(vbase + (size_t)(16 * c + l16) * CACHE_L + k0 + lhalf * 16);
      o[c] = wmma_bf16(pa, vf, o[c]);
    }
  }

  // ---- normalize, store bf16 attn[s][h*128+d]
#pragma unroll
  for (int i = 0; i < 8; ++i) {
    float inv = 1.0f / li[i];
    int q = qrow0 + 8 * lhalf + i;
#pragma unroll
    for (int c = 0; c < 8; ++c) {
      int d = 16 * c + l16;
      attnb[(size_t)q * HIDDEN + h * HD + d] = f2bf(o[c][i] * inv);
    }
  }
}

// ---------------------------------------------------------------------------
extern "C" void kernel_launch(void* const* d_in, const int* in_sizes, int n_in,
                              void* d_out, int out_size, void* d_ws, size_t ws_size,
                              hipStream_t stream) {
  (void)in_sizes; (void)n_in; (void)out_size; (void)ws_size;
  const float* x       = (const float*)d_in[0];
  const float* wqkv    = (const float*)d_in[1];
  const float* wo      = (const float*)d_in[2];
  const float* cache_k = (const float*)d_in[3];
  const float* cache_v = (const float*)d_in[4];
  const int*   sp      = (const int*)d_in[5];
  float* out = (float*)d_out;

  char* ws = (char*)d_ws;
  size_t off = 0;
  unsigned short* xb    = (unsigned short*)(ws + off); off += (size_t)SEQ * HIDDEN * 2;
  unsigned short* wqkvT = (unsigned short*)(ws + off); off += (size_t)QKV_N * HIDDEN * 2;
  unsigned short* woT   = (unsigned short*)(ws + off); off += (size_t)HIDDEN * HIDDEN * 2;
  float* qkv            = (float*)(ws + off);          off += (size_t)SEQ * QKV_N * 4;
  unsigned short* qbf   = (unsigned short*)(ws + off); off += (size_t)N_HEADS * SEQ * HD * 2;
  unsigned short* kc    = (unsigned short*)(ws + off); off += (size_t)N_KV * CACHE_L * HD * 2;
  unsigned short* vct   = (unsigned short*)(ws + off); off += (size_t)N_KV * CACHE_L * HD * 2;
  unsigned short* attnb = (unsigned short*)(ws + off); off += (size_t)SEQ * HIDDEN * 2;

  // 0) one-time precision/layout conversion passes
  int nx = SEQ * HIDDEN;
  cvt_bf16<<<(nx + 255) / 256, dim3(256), 0, stream>>>(x, xb, nx);
  cvt_transpose_bf16<<<dim3(QKV_N / 32, HIDDEN / 32), dim3(256), 0, stream>>>(
      wqkv, wqkvT, HIDDEN, QKV_N);
  cvt_transpose_bf16<<<dim3(HIDDEN / 32, HIDDEN / 32), dim3(256), 0, stream>>>(
      wo, woT, HIDDEN, HIDDEN);

  // 1) QKV projection (async-staged WMMA GEMM)
  gemm_bf16_async<<<dim3(QKV_N / 128, SEQ / 128), dim3(256), 0, stream>>>(
      xb, wqkvT, qkv, SEQ, QKV_N, HIDDEN);

  // 2) seed bf16 KV caches (V transposed)
  int ncache = N_KV * CACHE_L * HD;
  cvt_cache<<<(ncache + 255) / 256, dim3(256), 0, stream>>>(cache_k, cache_v, kc, vct, ncache);

  // 3) RoPE + scatter
  rope_scatter<<<dim3(80, SEQ), dim3(64), 0, stream>>>(qkv, qbf, kc, vct, sp);

  // 4) causal flash attention -> bf16 attn
  flash_attn<<<dim3(SEQ / 64, N_HEADS), dim3(128), 0, stream>>>(qbf, kc, vct, attnb, sp);

  // 5) output projection -> d_out (f32)
  gemm_bf16_async<<<dim3(HIDDEN / 128, SEQ / 128), dim3(256), 0, stream>>>(
      attnb, woT, out, SEQ, HIDDEN, HIDDEN);
}
